// SNNFabricModel_78426102825462
// MI455X (gfx1250) — matrix operand order repaired and checked by reading, not compiled, not run on hardware
//
#include <hip/hip_runtime.h>
#include <hip/hip_bf16.h>

// ---------------------------------------------------------------------------
// SNN recurrent network on MI455X (gfx1250).
//   B=32, T=256, N_IN=512, N_H=2048, N_OUT=512, alpha=0.9, thresh=1.0
// Strategy:
//   * input LIF precomputed (depends only on x) -> s_in[t] f16 in d_ws
//   * weights converted once to f16, transposed to [N][K] for contiguous
//     per-lane WMMA B-fragment loads (all weights ~12MB -> L2 resident)
//   * persistent kernel: 16 WGs x 256 thr; each wave owns two 16x16 tiles of
//     the hidden layer; v_h lives in the WMMA accumulator VGPRs across steps
//   * one global sync per step: atomic counter barrier + s_cluster_barrier
//   * after the sync, the full spike plane s_h(t) (32x2048 f16 = 128KB) is
//     pulled into LDS with TENSOR_LOAD_TO_LDS (TDM) + s_wait_tensorcnt; the
//     staged copy feeds BOTH the output-layer matmul of step t and the
//     recurrent matmul of step t+1 (A-fragments become ds_load_b128)
// ---------------------------------------------------------------------------

#define ALPHA   0.9f
#define BATCH   32
#define TSTEPS  256
#define N_IN    512
#define N_H     2048
#define N_OUT   512
#define NWG     16
#define NTH     256

typedef __attribute__((ext_vector_type(16))) _Float16 v16h;
typedef __attribute__((ext_vector_type(8)))  _Float16 v8h;
typedef __attribute__((ext_vector_type(8)))  float    v8f;
typedef unsigned int u32x4 __attribute__((ext_vector_type(4)));
typedef int          i32x4 __attribute__((ext_vector_type(4)));
typedef int          i32x8 __attribute__((ext_vector_type(8)));

// ---- d_ws layout (in _Float16 elements) -----------------------------------
#define OFF_W1T  ((size_t)0)                                   // [N_H][N_IN]
#define OFF_WRT  (OFF_W1T + (size_t)N_H * N_IN)                // [N_H][N_H]
#define OFF_W2T  (OFF_WRT + (size_t)N_H * N_H)                 // [N_OUT][N_H]
#define OFF_SIN  (OFF_W2T + (size_t)N_OUT * N_H)               // [T][B][N_IN]
#define OFF_SH   (OFF_SIN + (size_t)TSTEPS * BATCH * N_IN)     // 2*[B][N_H]
#define OFF_END  (OFF_SH  + (size_t)2 * BATCH * N_H)           // counter after

// ---------------------------------------------------------------------------
// WMMA helpers
// ---------------------------------------------------------------------------
__device__ __forceinline__ v8f wmma_f16(v16h a, v16h b, v8f c) {
  return __builtin_amdgcn_wmma_f32_16x16x32_f16(
      /*neg_a=*/false, a, /*neg_b=*/false, b,
      /*c_mod=*/(short)0, c, /*reuse_a=*/false, /*reuse_b=*/false);
}

// A fragment (16x32, M x K), matrix stored row-major [.., ld] f16.
// Contiguous k order (not the ISA table order): lanes 0-15 hold M=lane, K k..k+15;
//             lanes 16-31 hold M=lane-16, K k+16..k+31.
// Works for global or LDS (generic pointer; addrspace inferred).
__device__ __forceinline__ v16h load_frag_a(const _Float16* base, int ld,
                                            int m_base, int k) {
  const int lane = threadIdx.x & 31;
  const int r = lane & 15, hi = lane >> 4;
  const _Float16* p = base + (size_t)(m_base + r) * ld + k + hi * 16;
  v8h lo = *(const v8h*)p;
  v8h hf = *(const v8h*)(p + 8);
  v16h out;
#pragma unroll
  for (int i = 0; i < 8; ++i) { out[i] = lo[i]; out[i + 8] = hf[i]; }
  return out;
}

// B fragment (32x16, K x N), weight stored transposed [N][K] f16 (row = n).
// Contiguous k order, same as A (not ISA table order): lanes 0-15 hold N=lane, K k..k+15; lanes 16-31 K k+16..k+31.
__device__ __forceinline__ v16h load_frag_b(const _Float16* wt, int ld,
                                            int n_base, int k) {
  const int lane = threadIdx.x & 31;
  const int c = lane & 15, hi = lane >> 4;
  const _Float16* p = wt + (size_t)(n_base + c) * ld + k + hi * 16;
  return *(const v16h*)p;  // 32B contiguous, 32B aligned
}

// ---------------------------------------------------------------------------
// TDM: stage spike plane [32][2048] f16 (128KB) global -> LDS.
// D# per CDNA5 ISA ch.8: group0 = {count/lds_addr/global_addr/type},
// group1 = {mask/data_size/dims/strides}; 2D tile, data_size=2B,
// tensor_dim0=tile_dim0=2048, tensor_dim1=tile_dim1=32, stride0=2048.
// ---------------------------------------------------------------------------
__device__ __forceinline__ void tdm_stage_sh(const _Float16* gsrc,
                                             unsigned lds_byte_addr) {
  unsigned long long ga = (unsigned long long)(uintptr_t)gsrc;
  u32x4 g0;
  g0.x = 1u;                                        // count=1, load, user
  g0.y = lds_byte_addr;                             // [63:32] lds_addr
  g0.z = (unsigned)(ga & 0xFFFFFFFFu);              // [95:64] gaddr lo
  g0.w = (unsigned)((ga >> 32) & 0x1FFFFFFu)        // [120:96] gaddr hi
         | (2u << 30);                              // [127:126] type=2
  i32x8 g1;
  g1[0] = (int)(1u << 16);          // wg_mask=0, data_size=1 (2 bytes)
  g1[1] = (int)(2048u << 16);       // atomic_bar=0 | tensor_dim0[15:0]
  g1[2] = (int)(32u << 16);         // tensor_dim0[31:16]=0 | tensor_dim1 lo
  g1[3] = (int)(2048u << 16);       // tensor_dim1 hi=0 | tile_dim0=2048
  g1[4] = 32;                       // tile_dim1=32 | tile_dim2=0
  g1[5] = 2048;                     // tensor_dim0_stride lo32
  g1[6] = 0;                        // stride0 hi | stride1 lo
  g1[7] = 0;                        // stride1 hi
  i32x4 gz = {0, 0, 0, 0};
#if __clang_major__ >= 23
  i32x8 gz8 = {0, 0, 0, 0, 0, 0, 0, 0};
  __builtin_amdgcn_tensor_load_to_lds(g0, g1, gz, gz, gz8, 0);
#else
  __builtin_amdgcn_tensor_load_to_lds(g0, g1, gz, gz, 0);
#endif
}

// ---------------------------------------------------------------------------
// One-time prep: f32->f16 transpose of W1/Wr/W2, zero spike buffers + counter
// ---------------------------------------------------------------------------
__global__ void k_prep(const float* __restrict__ W1,
                       const float* __restrict__ Wr,
                       const float* __restrict__ W2,
                       _Float16* __restrict__ ws,
                       unsigned* __restrict__ ctr) {
  const size_t n_w1 = (size_t)N_H * N_IN;
  const size_t n_wr = (size_t)N_H * N_H;
  const size_t n_w2 = (size_t)N_OUT * N_H;
  const size_t n_sh = (size_t)2 * BATCH * N_H;
  size_t idx = (size_t)blockIdx.x * blockDim.x + threadIdx.x;
  if (idx == 0) *ctr = 0u;
  if (idx < n_w1) {                               // W1t[n][k] = W1[k][n]
    size_t n = idx >> 9, k = idx & (N_IN - 1);
    ws[OFF_W1T + idx] = (_Float16)W1[k * N_H + n];
  } else if (idx < n_w1 + n_wr) {                 // Wrt[n][k] = Wr[k][n]
    size_t l = idx - n_w1;
    size_t n = l >> 11, k = l & (N_H - 1);
    ws[OFF_WRT + l] = (_Float16)Wr[k * N_H + n];
  } else if (idx < n_w1 + n_wr + n_w2) {          // W2t[n][k] = W2[k][n]
    size_t l = idx - n_w1 - n_wr;
    size_t n = l >> 11, k = l & (N_H - 1);
    ws[OFF_W2T + l] = (_Float16)W2[k * N_OUT + n];
  } else if (idx < n_w1 + n_wr + n_w2 + n_sh) {   // zero s_h double buffer
    ws[OFF_SH + (idx - n_w1 - n_wr - n_w2)] = (_Float16)0.0f;
  }
}

// ---------------------------------------------------------------------------
// Input-population LIF: depends only on x -> precompute all T spike planes
// ---------------------------------------------------------------------------
__global__ void k_input_spikes(const float* __restrict__ x,
                               _Float16* __restrict__ sin_all) {
  const int idx = blockIdx.x * blockDim.x + threadIdx.x;  // b*512 + i
  const int b = idx >> 9;
  const int i = idx & (N_IN - 1);
  float v = 0.0f;
  for (int t = 0; t < TSTEPS; ++t) {
    v = ALPHA * v + x[((size_t)b * TSTEPS + t) * N_IN + i];
    float s = (v - 1.0f >= 0.0f) ? 1.0f : 0.0f;
    v *= (1.0f - s);
    sin_all[(size_t)t * (BATCH * N_IN) + idx] = (_Float16)s;
  }
}

// ---------------------------------------------------------------------------
// Per-step grid sync: atomic counter (functional) + cluster barrier (CDNA5)
// ---------------------------------------------------------------------------
__device__ __forceinline__ void gsync(unsigned* ctr, int t) {
  __threadfence();
  __syncthreads();
  if (threadIdx.x == 0) {
    atomicAdd(ctr, 1u);
    const unsigned target = (unsigned)(NWG * (t + 1));
    volatile unsigned* vc = (volatile unsigned*)ctr;
    while (*vc < target) { __builtin_amdgcn_s_sleep(2); }
    // cluster split-barrier (s_barrier_signal -3 / s_barrier_wait -3);
    // hardware NOP when not dispatched as a cluster
    __builtin_amdgcn_s_cluster_barrier();
  }
  __syncthreads();
  __threadfence();
}

// ---------------------------------------------------------------------------
// Persistent recurrent SNN kernel: 16 WGs x 256 threads (8 waves)
// ---------------------------------------------------------------------------
__global__ void __launch_bounds__(NTH, 1)
k_snn(const _Float16* __restrict__ W1t, const _Float16* __restrict__ Wrt,
      const _Float16* __restrict__ W2t, const _Float16* __restrict__ sin_all,
      _Float16* __restrict__ shbuf, unsigned* __restrict__ ctr,
      float* __restrict__ out) {
  __shared__ _Float16 lds_sh[BATCH * N_H];   // 128KB staged spike plane

  const int wg   = blockIdx.x;          // 0..15
  const int wv   = threadIdx.x >> 5;    // 0..7
  const int lane = threadIdx.x & 31;
  const int nbh  = wg * 128 + wv * 16;  // hidden column base of this wave
  const int om   = wv & 1;              // output M-tile (waves 0-3)
  const int on   = wv >> 1;             // output N-tile (waves 0-3)
  const int nbo  = wg * 32 + on * 16;   // output column base

  v8f vh0 = {}, vh1 = {};               // hidden membrane (2 tiles, in VGPRs)
  v8f vout = {}, accout = {};           // output membrane + spike accumulator

  _Float16* sh0 = shbuf;
  _Float16* sh1 = shbuf + (size_t)BATCH * N_H;

  // s_h(-1) = 0 : zero the LDS staging buffer
  {
    v8h z = {};
    v8h* l8 = (v8h*)lds_sh;
    for (int i = threadIdx.x; i < (BATCH * N_H) / 8; i += NTH) l8[i] = z;
  }
  __syncthreads();
  const unsigned lds_base = (unsigned)(uintptr_t)(void*)lds_sh;

  for (int t = 0; t < TSTEPS; ++t) {
    _Float16* shCur = (t & 1) ? sh0 : sh1;   // global exchange buffer
    const _Float16* sint = sin_all + (size_t)t * (BATCH * N_IN);

    v8f ih0 = {}, ih1 = {};

    // ---- feedforward: s_in (32x512) @ W1[:, slice]  (A from global) ------
    for (int k = 0; k < N_IN; k += 32) {
      v16h a0 = load_frag_a(sint, N_IN, 0,  k);
      v16h a1 = load_frag_a(sint, N_IN, 16, k);
      v16h bf = load_frag_b(W1t, N_IN, nbh, k);
      ih0 = wmma_f16(a0, bf, ih0);
      ih1 = wmma_f16(a1, bf, ih1);
    }
    // ---- recurrent: s_h(t-1) (32x2048) @ Wr[:, slice]  (A from LDS) ------
    for (int k = 0; k < N_H; k += 32) {
      v16h a0 = load_frag_a(lds_sh, N_H, 0,  k);
      v16h a1 = load_frag_a(lds_sh, N_H, 16, k);
      v16h bf = load_frag_b(Wrt, N_H, nbh, k);
      // keep streamed weight rows warm (global_prefetch_b8)
      __builtin_prefetch((const void*)(Wrt + (size_t)(nbh + (lane & 15)) * N_H
                                       + k + 64), 0, 1);
      ih0 = wmma_f16(a0, bf, ih0);
      ih1 = wmma_f16(a1, bf, ih1);
    }

    // ---- hidden LIF in accumulator registers; scatter spikes (f16) -------
    {
      const int col   = nbh + (lane & 15);
      const int rbase = (lane >> 4) * 8;     // C/D: VGPR j -> M = j + 8*hi
#pragma unroll
      for (int j = 0; j < 8; ++j) {
        const int brow = rbase + j;
        float v = ALPHA * vh0[j] + ih0[j];
        float s = (v - 1.0f >= 0.0f) ? 1.0f : 0.0f;
        vh0[j] = v * (1.0f - s);
        shCur[(size_t)brow * N_H + col] = (_Float16)s;

        v = ALPHA * vh1[j] + ih1[j];
        s = (v - 1.0f >= 0.0f) ? 1.0f : 0.0f;
        vh1[j] = v * (1.0f - s);
        shCur[(size_t)(16 + brow) * N_H + col] = (_Float16)s;
      }
    }

    gsync(ctr, t);  // spikes of step t visible device-wide; LDS reads done

    // ---- TDM: pull full s_h(t) plane into LDS (serves output layer of
    //      step t and recurrent layer of step t+1) -------------------------
    if (threadIdx.x == 0) {
      tdm_stage_sh(shCur, lds_base);
      __builtin_amdgcn_s_wait_tensorcnt(0);
    }
    __syncthreads();

    // ---- output layer: s_h(t) (32x2048) @ W2[:, slice]; waves 0-3 --------
    if (wv < 4) {
      v8f io = {};
      for (int k = 0; k < N_H; k += 32) {
        v16h a  = load_frag_a(lds_sh, N_H, om * 16, k);
        v16h bf = load_frag_b(W2t, N_H, nbo, k);
        io = wmma_f16(a, bf, io);
      }
#pragma unroll
      for (int j = 0; j < 8; ++j) {
        float v = ALPHA * vout[j] + io[j];
        float s = (v - 1.0f >= 0.0f) ? 1.0f : 0.0f;
        vout[j] = v * (1.0f - s);
        accout[j] += s;
      }
    }
  }

  // ---- write acc_out [32][512] f32 ---------------------------------------
  if (wv < 4) {
    const int col   = nbo + (lane & 15);
    const int rbase = om * 16 + (lane >> 4) * 8;
#pragma unroll
    for (int j = 0; j < 8; ++j)
      out[(size_t)(rbase + j) * N_OUT + col] = accout[j];
  }
}

// ---------------------------------------------------------------------------
extern "C" void kernel_launch(void* const* d_in, const int* in_sizes, int n_in,
                              void* d_out, int out_size, void* d_ws,
                              size_t ws_size, hipStream_t stream) {
  const float* x  = (const float*)d_in[0];   // [32][256][512]
  const float* W1 = (const float*)d_in[1];   // [512][2048]
  const float* Wr = (const float*)d_in[2];   // [2048][2048]
  const float* W2 = (const float*)d_in[3];   // [2048][512]
  float* out = (float*)d_out;                // [32][512]
  _Float16* ws = (_Float16*)d_ws;            // needs ~20.3 MB
  unsigned* ctr = (unsigned*)(ws + OFF_END); // 4B counter after f16 arena

  const size_t prep_elems = OFF_SH + (size_t)2 * BATCH * N_H;  // weights+sh
  const unsigned prep_blocks = (unsigned)((prep_elems + NTH - 1) / NTH);
  k_prep<<<prep_blocks, NTH, 0, stream>>>(W1, Wr, W2, ws, ctr);

  k_input_spikes<<<(BATCH * N_IN) / NTH, NTH, 0, stream>>>(x, ws + OFF_SIN);

  k_snn<<<NWG, NTH, 0, stream>>>(ws + OFF_W1T, ws + OFF_WRT, ws + OFF_W2T,
                                 ws + OFF_SIN, ws + OFF_SH, ctr, out);
}
